// TTS_54425825575315
// MI455X (gfx1250) — compile-verified, matrix operand order checked by reading, not run on hardware
//
#include <hip/hip_runtime.h>

// ---------------------------------------------------------------------------
// GMM-attention LSTM decoder for MI455X (gfx1250, wave32, WMMA).
// B=16, T=1024, D=1024, S=512, M=16.
// Two fused kernels per step (launch-overhead bound otherwise):
//   k_gates_lstm : gates(16x4096) = X(16x3072) @ Wcat^T via
//                  v_wmma_f32_16x16x32_bf16, gate-interleaved column packing,
//                  LDS accumulator exchange, LSTM elementwise in-block.
//   k_attn_ctx   : phi(16x48) WMMA (redundant per block), GMM window into
//                  LDS, context GEMV slice per block (fp32).
// X is ping-ponged between steps to avoid intra-dispatch read/write races.
// ---------------------------------------------------------------------------

typedef __bf16 bf16_t;
typedef bf16_t v16bf __attribute__((ext_vector_type(16)));
typedef bf16_t v8bf  __attribute__((ext_vector_type(8)));
typedef float  v8f   __attribute__((ext_vector_type(8)));

#define BATCH 16
#define TLEN  1024
#define DDIM  1024
#define SLEN  512
#define KDIM  3072            // 2*D (x,ctx) + D (h)
#define NGATE 4096            // 4*D
#define KT_GATES (KDIM / 32)  // 96
#define KT_ATTN  (DDIM / 32)  // 32

__device__ __forceinline__ float sigf(float x) {
    return 1.0f / (1.0f + __expf(-x));
}

// ---------------- one-time weight packing into WMMA B-fragment order -------
// Packed tile = 16(N) x 32(K) bf16, 512 elems. Lane l holds column n'=l&15;
// its 16 elems cover K' = {kh..kh+7, kh+16..kh+23}, kh = (l<16)?0:8.
// Packed column pc (0..4095) is gate-interleaved so one block owns a full
// (i,f,g,o) set for 32 consecutive d:
//   bk = pc>>7, local = pc&127, d = bk*32 + (local>>2), gate = local&3
//   logical column = gate*1024 + d
__global__ __launch_bounds__(256) void k_pack_gates_w(
    const float* __restrict__ W_ih, const float* __restrict__ W_hh,
    bf16_t* __restrict__ Wp)
{
    size_t idx = (size_t)blockIdx.x * 256 + threadIdx.x; // 4096*3072 total
    int e  = (int)(idx & 15);
    int l  = (int)((idx >> 4) & 31);
    size_t tile = idx >> 9;
    int kt = (int)(tile % KT_GATES);
    int nt = (int)(tile / KT_GATES);
    int pc = nt * 16 + (l & 15);                 // packed column
    int bk = pc >> 7, local = pc & 127;
    int n  = (local & 3) * 1024 + bk * 32 + (local >> 2);  // logical column
    int kh = (l < 16) ? 0 : 8;
    int kp = (e < 8) ? (kh + e) : (kh + 8 + e);
    int k  = kt * 32 + kp;
    float v = (k < 2048) ? W_ih[(size_t)n * 2048 + k]
                         : W_hh[(size_t)n * 1024 + (k - 2048)];
    Wp[idx] = (bf16_t)v;
}

__global__ __launch_bounds__(256) void k_pack_attn_w(
    const float* __restrict__ Wg_w, bf16_t* __restrict__ Wgp)
{
    size_t idx = (size_t)blockIdx.x * 256 + threadIdx.x; // 48*1024 total
    int e  = (int)(idx & 15);
    int l  = (int)((idx >> 4) & 31);
    size_t tile = idx >> 9;
    int kt = (int)(tile % KT_ATTN);
    int nt = (int)(tile / KT_ATTN);
    int n  = nt * 16 + (l & 15);
    int kh = (l < 16) ? 0 : 8;
    int kp = (e < 8) ? (kh + e) : (kh + 8 + e);
    int k  = kt * 32 + kp;
    Wgp[idx] = (bf16_t)Wg_w[(size_t)n * 1024 + k];
}

// ---------------- init: X0 = [x_0 | 0 | 0], c = 0 --------------------------
__global__ __launch_bounds__(256) void k_init(
    const float* __restrict__ input_h_c, bf16_t* __restrict__ X,
    float* __restrict__ c)
{
    int idx = blockIdx.x * 256 + threadIdx.x;   // 16384
    int b = idx >> 10, d = idx & 1023;
    X[b * KDIM + d]          = (bf16_t)input_h_c[(size_t)b * TLEN * DDIM + d];
    X[b * KDIM + 1024 + d]   = (bf16_t)0.0f;
    X[b * KDIM + 2048 + d]   = (bf16_t)0.0f;
    c[idx] = 0.0f;
}

// ---------------- fused gates GEMM + LSTM ----------------------------------
// 32 blocks x 8 waves. Block bk owns packed columns [bk*128, bk*128+128)
// = gates (i,f,g,o) for d in [bk*32, bk*32+32), all 16 batches.
__global__ __launch_bounds__(256) void k_gates_lstm(
    const bf16_t* __restrict__ Xin, bf16_t* __restrict__ Xout,
    const bf16_t* __restrict__ Wp, const float* __restrict__ b_ih,
    const float* __restrict__ b_hh, float* __restrict__ c,
    const float* __restrict__ input_h_c, int t)
{
    __shared__ float g_lds[BATCH][128];

    int lane  = threadIdx.x & 31;
    int wave  = threadIdx.x >> 5;
    int ntile = blockIdx.x * 8 + wave;          // 0..255
    int row   = lane & 15;
    int kh    = (lane < 16) ? 0 : 8;
    const bf16_t* xrow  = Xin + row * KDIM + kh;
    const bf16_t* wbase = Wp + (size_t)ntile * KT_GATES * 512 + lane * 16;

    v8f acc = {};
    #pragma unroll 4
    for (int kt = 0; kt < KT_GATES; ++kt) {
        __builtin_prefetch(wbase + (kt + 6) * 512, 0, 3);
        union { v16bf v; v8bf h[2]; } a;
        a.h[0] = *(const v8bf*)(xrow + kt * 32);
        a.h[1] = *(const v8bf*)(xrow + kt * 32 + 16);
        v16bf bfrag = *(const v16bf*)(wbase + (size_t)kt * 512);
        acc = __builtin_amdgcn_wmma_f32_16x16x32_bf16(
            false, a.v, false, bfrag, (short)0, acc, false, false);
    }
    int mbase = (lane >> 4) * 8;
    int ncol  = wave * 16 + (lane & 15);        // local packed column 0..127
    #pragma unroll
    for (int r = 0; r < 8; ++r)
        g_lds[mbase + r][ncol] = acc[r];
    __syncthreads();

    // LSTM elementwise: 16 batches x 32 d per block = 512 pairs, 2/thread
    int bk = blockIdx.x;
    for (int p = threadIdx.x; p < BATCH * 32; p += 256) {
        int b = p >> 5, dl = p & 31;
        int d = bk * 32 + dl;
        float i_ = g_lds[b][dl * 4 + 0] + b_ih[d]        + b_hh[d];
        float f_ = g_lds[b][dl * 4 + 1] + b_ih[1024 + d] + b_hh[1024 + d];
        float g_ = g_lds[b][dl * 4 + 2] + b_ih[2048 + d] + b_hh[2048 + d];
        float o_ = g_lds[b][dl * 4 + 3] + b_ih[3072 + d] + b_hh[3072 + d];
        float cn = sigf(f_) * c[b * DDIM + d] + sigf(i_) * tanhf(g_);
        c[b * DDIM + d] = cn;
        float h = sigf(o_) * tanhf(cn);
        Xout[b * KDIM + 2048 + d] = (bf16_t)h;
        if (t + 1 < TLEN)
            Xout[b * KDIM + d] = (bf16_t)
                input_h_c[(size_t)b * TLEN * DDIM + (size_t)(t + 1) * DDIM + d];
    }
}

// ---------------- fused attention + context --------------------------------
// 64 blocks: block = b*4 + quarter. phi WMMA is recomputed per block (cheap),
// GMM window w[b,:] built in LDS, then a 256-wide context GEMV slice.
__global__ __launch_bounds__(256) void k_attn_ctx(
    bf16_t* __restrict__ X, const bf16_t* __restrict__ Wgp,
    const float* __restrict__ Wg_b, const int* __restrict__ lens,
    const float* __restrict__ memory, float* __restrict__ ctx_out,
    float* __restrict__ align_out, float* __restrict__ term_out,
    int t, int is_last)
{
    __shared__ float phi[BATCH][48];
    __shared__ float s_alpha[16], s_ksi[16], s_invb[16];
    __shared__ float wbuf[SLEN];

    int b       = blockIdx.x >> 2;
    int quarter = blockIdx.x & 3;
    int lane = threadIdx.x & 31;
    int wave = threadIdx.x >> 5;

    if (wave < 3) {                              // phi = h @ Wg^T + Wg_b
        int row = lane & 15;
        int kh  = (lane < 16) ? 0 : 8;
        const bf16_t* xrow  = X + row * KDIM + 2048 + kh;   // h section
        const bf16_t* wbase = Wgp + (size_t)wave * KT_ATTN * 512 + lane * 16;
        v8f acc = {};
        #pragma unroll 4
        for (int kt = 0; kt < KT_ATTN; ++kt) {
            union { v16bf v; v8bf h[2]; } a;
            a.h[0] = *(const v8bf*)(xrow + kt * 32);
            a.h[1] = *(const v8bf*)(xrow + kt * 32 + 16);
            v16bf bfrag = *(const v16bf*)(wbase + (size_t)kt * 512);
            acc = __builtin_amdgcn_wmma_f32_16x16x32_bf16(
                false, a.v, false, bfrag, (short)0, acc, false, false);
        }
        int mbase = (lane >> 4) * 8;
        int n     = wave * 16 + (lane & 15);
        #pragma unroll
        for (int r = 0; r < 8; ++r)
            phi[mbase + r][n] = acc[r] + Wg_b[n];
    }
    __syncthreads();

    if (threadIdx.x == 0) {                      // params for this block's b
        float mx = -3.4e38f;
        #pragma unroll
        for (int m = 0; m < 16; ++m) mx = fmaxf(mx, phi[b][32 + m]);
        float se = 0.0f;
        #pragma unroll
        for (int m = 0; m < 16; ++m) {
            float e = __expf(phi[b][32 + m] - mx);
            s_alpha[m] = e; se += e;
        }
        float inv = 1.0f / se;
        #pragma unroll
        for (int m = 0; m < 16; ++m) {
            s_alpha[m] *= inv;
            s_ksi[m]  = __expf(phi[b][m]);
            s_invb[m] = __expf(-phi[b][16 + m]);   // 1/beta
        }
    }
    __syncthreads();

    for (int s = threadIdx.x; s < SLEN; s += 256) {
        float w = 0.0f, rsum = 0.0f;
        float sf = (float)s;
        #pragma unroll
        for (int m = 0; m < 16; ++m) {
            float ib = s_invb[m];
            float z  = (sf - s_ksi[m]) * ib;
            float hb = 0.5f * ib;
            float FR = sigf(z + hb);
            float FL = sigf(z - hb);
            float al = s_alpha[m];
            w    += al * (FR - FL);
            rsum += al * FR;
        }
        wbuf[s] = w;
        if (quarter == 0) {
            align_out[(size_t)b * TLEN * SLEN + (size_t)t * SLEN + s] = w;
            if (is_last && s == lens[b] - 1)
                term_out[b] = 1.0f - rsum;
        }
    }
    __syncthreads();

    // ctx[b,d] = sum_s w[b,s] * memory[b,s,d], d slice of 256 per block
    int d = quarter * 256 + threadIdx.x;
    const float* mem = memory + (size_t)b * SLEN * DDIM + d;
    float acc = 0.0f;
    #pragma unroll 8
    for (int s = 0; s < SLEN; ++s)
        acc = fmaf(wbuf[s], mem[(size_t)s * DDIM], acc);
    ctx_out[b * DDIM + d] = acc;                 // (b,0,d): last step survives
    X[b * KDIM + 1024 + d] = (bf16_t)acc;        // bf16 for next step's GEMM
}

// ---------------------------------------------------------------------------
extern "C" void kernel_launch(void* const* d_in, const int* in_sizes, int n_in,
                              void* d_out, int out_size, void* d_ws, size_t ws_size,
                              hipStream_t stream)
{
    const float* input_h_c = (const float*)d_in[0];
    const float* memory    = (const float*)d_in[1];
    const int*   lens      = (const int*)  d_in[2];
    const float* W_ih      = (const float*)d_in[3];
    const float* W_hh      = (const float*)d_in[4];
    const float* b_ih      = (const float*)d_in[5];
    const float* b_hh      = (const float*)d_in[6];
    const float* Wg_w      = (const float*)d_in[7];
    const float* Wg_b      = (const float*)d_in[8];

    float* ctx_out   = (float*)d_out;                       // 16*1024
    float* align_out = ctx_out + BATCH * DDIM;              // 16*1024*512
    float* term_out  = align_out + (size_t)BATCH * TLEN * SLEN; // 16

    // workspace carve-up
    char* ws = (char*)d_ws;
    size_t off = 0;
    bf16_t* Wp  = (bf16_t*)(ws + off); off += (size_t)NGATE * KDIM * 2;   // 24 MB
    bf16_t* Wgp = (bf16_t*)(ws + off); off += 48 * 1024 * 2;              // 96 KB
    bf16_t* X0  = (bf16_t*)(ws + off); off += (size_t)BATCH * KDIM * 2;   // 96 KB
    bf16_t* X1  = (bf16_t*)(ws + off); off += (size_t)BATCH * KDIM * 2;   // 96 KB
    float*  cst = (float*) (ws + off); off += (size_t)BATCH * DDIM * 4;   // 64 KB

    // one-time packs + init (re-run every call: deterministic)
    k_pack_gates_w<<<(NGATE * KDIM) / 256, 256, 0, stream>>>(W_ih, W_hh, Wp);
    k_pack_attn_w <<<(48 * 1024) / 256,    256, 0, stream>>>(Wg_w, Wgp);
    k_init        <<<64, 256, 0, stream>>>(input_h_c, X0, cst);

    for (int t = 0; t < TLEN; ++t) {
        bf16_t* Xin  = (t & 1) ? X1 : X0;
        bf16_t* Xout = (t & 1) ? X0 : X1;
        k_gates_lstm<<<32, 256, 0, stream>>>(Xin, Xout, Wp, b_ih, b_hh, cst,
                                             input_h_c, t);
        k_attn_ctx  <<<64, 256, 0, stream>>>(Xout, Wgp, Wg_b, lens, memory,
                                             ctx_out, align_out, term_out,
                                             t, t == TLEN - 1 ? 1 : 0);
    }
}
